// ConvBNReLU_67053029425821
// MI455X (gfx1250) — compile-verified, matrix operand order checked by reading, not run on hardware
//
#include <hip/hip_runtime.h>
#include <hip/hip_bf16.h>
#include <stdint.h>

typedef __attribute__((ext_vector_type(16))) __bf16 v16bf;
typedef __attribute__((ext_vector_type(8)))  float  v8f;

#define NPTS        (1 << 20)
#define CIN         32
#define COUT        64
#define KOFF        9
#define NTILES      (NPTS / 16)          // 65536 row-tiles of 16 points
#define MAIN_BLOCKS 1024
#define THREADS     256
#define WPB         (THREADS / 32)       // 8 waves per block (wave32)
#define EPS         1e-5f

// Workspace layout (in floats):
//   [0 .. MAIN_BLOCKS*128)   : per-block partials (64 sums, then 64 sumsq)
//   [WS_SCALE_F .. +64)      : scale  (gamma * rsqrt(var+eps))
//   [WS_SHIFT_F .. +64)      : shift  (beta - mean*scale)
//   [WS_WOP_F   .. )         : bf16 W in B-operand layout [K][4 tiles][32 lanes][16 elems]
#define WS_SCALE_F  (MAIN_BLOCKS * 128)
#define WS_SHIFT_F  (WS_SCALE_F + 64)
#define WS_WOP_F    (WS_SHIFT_F + 64)
#define WOP_ELEMS   (KOFF * 4 * 32 * 16) // 18432 bf16 = 36864 bytes

// ---------------------------------------------------------------------------
// Kernel 1: lay out W (f32 [9][32][64]) as bf16 B-operands for
// v_wmma_f32_16x16x32_bf16.  B operand per lane: 16 bf16.  Lane l (half h =
// l>>4, col n = l&15), element j maps to K index kk = h*8 + j + (j>=8 ? 8:0),
// matching the ISA 16-bit A/B K-packing (K 0..7 / 16..23 in half 0, 8..15 /
// 24..31 in half 1).
// ---------------------------------------------------------------------------
__global__ __launch_bounds__(THREADS)
void prep_kernel(const float* __restrict__ W, float* __restrict__ ws) {
    __bf16* wop = (__bf16*)(ws + WS_WOP_F);
    for (int i = threadIdx.x; i < WOP_ELEMS; i += THREADS) {
        int j   = i & 15;
        int l   = (i >> 4) & 31;
        int t   = (i >> 9) & 3;
        int k   = i >> 11;
        int h   = l >> 4;
        int n   = l & 15;
        int kk  = h * 8 + j + ((j >= 8) ? 8 : 0);
        wop[i] = (__bf16)W[(size_t)(k * CIN + kk) * COUT + t * 16 + n];
    }
}

// ---------------------------------------------------------------------------
// Kernel 2: gather + bf16 WMMA GEMM + per-block BN partial stats.
// One wave per 16-point tile; 4 cout tiles of 16; K = 9 offsets x 32 = 288.
// k-loop kept rolled (#pragma unroll 1) so the four accumulators stay in
// fixed VGPR blocks across iterations -> no v_mov_b64 accumulator shuffling,
// and the WMMA->WMMA hazard slots are filled by the next iteration's
// independent gather/convert work.
// ---------------------------------------------------------------------------
__global__ __launch_bounds__(THREADS)
void main_kernel(const float* __restrict__ feats,
                 const int* __restrict__ nb_idx,
                 const unsigned char* __restrict__ nb_mask,
                 const float* __restrict__ ws_ro,
                 float* __restrict__ partial,
                 float* __restrict__ out) {
    __shared__ __align__(32) __bf16 sB[WOP_ELEMS];   // 36 KB
    __shared__ float sSum[WPB][2][COUT];             // 4 KB
    __shared__ float sSq [WPB][2][COUT];             // 4 KB

    const int tid = threadIdx.x;

    // Stage pre-laid-out bf16 W into LDS (2304 float4 = 36864 B).
    {
        const float4* src = (const float4*)(ws_ro + WS_WOP_F);
        float4*       dst = (float4*)sB;
        for (int i = tid; i < WOP_ELEMS * 2 / 16; i += THREADS) dst[i] = src[i];
    }
    __syncthreads();

    const int lane = tid & 31;
    const int wave = tid >> 5;
    const int h    = lane >> 4;   // half-wave
    const int m    = lane & 15;   // row within tile (A), col within tile (B/C/D)

    const v16bf* blane = (const v16bf*)sB + lane;  // stride 32 lanes per (k,t)

    float lsum[4] = {0.f, 0.f, 0.f, 0.f};
    float lsq [4] = {0.f, 0.f, 0.f, 0.f};

    const int gwave  = blockIdx.x * WPB + wave;
    const int stride = MAIN_BLOCKS * WPB;

    for (int tile = gwave; tile < NTILES; tile += stride) {
        const int p = tile * 16 + m;   // point index this lane gathers for
        v8f c0 = {}, c1 = {}, c2 = {}, c3 = {};

        // Software-pipelined idx/mask for offset 0.
        int   idx  = nb_idx[p];
        float mval = nb_mask[p] ? 1.0f : 0.0f;

#pragma unroll 1
        for (int k = 0; k < KOFF; ++k) {
            const float* row = feats + (size_t)idx * CIN + h * 8;

            // Two 8-float runs: K in [base, base+8) and [base+16, base+24).
            float4 r0 = *(const float4*)(row + 0);
            float4 r1 = *(const float4*)(row + 4);
            float4 r2 = *(const float4*)(row + 16);
            float4 r3 = *(const float4*)(row + 20);

            // Prefetch next offset's index/mask (independent of the WMMAs).
            int   nidx  = 0;
            float nmval = 0.0f;
            if (k + 1 < KOFF) {
                nidx  = nb_idx[(k + 1) * NPTS + p];
                nmval = nb_mask[(k + 1) * NPTS + p] ? 1.0f : 0.0f;
            }

            v16bf a;
            a[0]  = (__bf16)(r0.x * mval); a[1]  = (__bf16)(r0.y * mval);
            a[2]  = (__bf16)(r0.z * mval); a[3]  = (__bf16)(r0.w * mval);
            a[4]  = (__bf16)(r1.x * mval); a[5]  = (__bf16)(r1.y * mval);
            a[6]  = (__bf16)(r1.z * mval); a[7]  = (__bf16)(r1.w * mval);
            a[8]  = (__bf16)(r2.x * mval); a[9]  = (__bf16)(r2.y * mval);
            a[10] = (__bf16)(r2.z * mval); a[11] = (__bf16)(r2.w * mval);
            a[12] = (__bf16)(r3.x * mval); a[13] = (__bf16)(r3.y * mval);
            a[14] = (__bf16)(r3.z * mval); a[15] = (__bf16)(r3.w * mval);

            v16bf b0 = blane[(k * 4 + 0) * 32];
            v16bf b1 = blane[(k * 4 + 1) * 32];
            v16bf b2 = blane[(k * 4 + 2) * 32];
            v16bf b3 = blane[(k * 4 + 3) * 32];

            c0 = __builtin_amdgcn_wmma_f32_16x16x32_bf16(false, a, false, b0,
                                                         (short)0, c0, false, false);
            c1 = __builtin_amdgcn_wmma_f32_16x16x32_bf16(false, a, false, b1,
                                                         (short)0, c1, false, false);
            c2 = __builtin_amdgcn_wmma_f32_16x16x32_bf16(false, a, false, b2,
                                                         (short)0, c2, false, false);
            c3 = __builtin_amdgcn_wmma_f32_16x16x32_bf16(false, a, false, b3,
                                                         (short)0, c3, false, false);

            idx  = nidx;
            mval = nmval;
        }

        // Store: C/D layout -> lane holds col n=m; VGPR r holds row M = r + h*8.
        float* obase = out + (size_t)tile * 16 * COUT + m;
#pragma unroll
        for (int r = 0; r < 8; ++r) {
            const int M = r + h * 8;
            float v;
            v = c0[r]; obase[(size_t)M * COUT +  0] = v; lsum[0] += v; lsq[0] += v * v;
            v = c1[r]; obase[(size_t)M * COUT + 16] = v; lsum[1] += v; lsq[1] += v * v;
            v = c2[r]; obase[(size_t)M * COUT + 32] = v; lsum[2] += v; lsq[2] += v * v;
            v = c3[r]; obase[(size_t)M * COUT + 48] = v; lsum[3] += v; lsq[3] += v * v;
        }
    }

    // Deterministic fixed-order block reduction of BN partials.
#pragma unroll
    for (int t = 0; t < 4; ++t) {
        sSum[wave][h][t * 16 + m] = lsum[t];
        sSq [wave][h][t * 16 + m] = lsq[t];
    }
    __syncthreads();
    if (tid < 64) {
        float a = 0.f;
        for (int w = 0; w < WPB; ++w)
            for (int hh = 0; hh < 2; ++hh) a += sSum[w][hh][tid];
        partial[(size_t)blockIdx.x * 128 + tid] = a;
    } else if (tid < 128) {
        const int ch = tid - 64;
        float a = 0.f;
        for (int w = 0; w < WPB; ++w)
            for (int hh = 0; hh < 2; ++hh) a += sSq[w][hh][ch];
        partial[(size_t)blockIdx.x * 128 + tid] = a;
    }
}

// ---------------------------------------------------------------------------
// Kernel 3: reduce block partials -> scale/shift for fused BN.
// ---------------------------------------------------------------------------
__global__ __launch_bounds__(64)
void stats_kernel(float* __restrict__ ws,
                  const float* __restrict__ gamma,
                  const float* __restrict__ beta) {
    const int ch = threadIdx.x;   // 0..63
    float s = 0.f, sq = 0.f;
    for (int b = 0; b < MAIN_BLOCKS; ++b) {
        s  += ws[(size_t)b * 128 + ch];
        sq += ws[(size_t)b * 128 + 64 + ch];
    }
    const float invn = 1.0f / (float)NPTS;
    const float mean = s * invn;
    const float var  = sq * invn - mean * mean;
    const float inv  = rsqrtf(var + EPS);
    const float sc   = gamma[ch] * inv;
    ws[WS_SCALE_F + ch] = sc;
    ws[WS_SHIFT_F + ch] = beta[ch] - mean * sc;
}

// ---------------------------------------------------------------------------
// Kernel 4: y = clamp(scale*acc + shift, 0, 6), float4-vectorized in place.
// ---------------------------------------------------------------------------
__global__ __launch_bounds__(THREADS)
void norm_kernel(const float* __restrict__ ws, float* __restrict__ out) {
    __shared__ float ss[128];     // [0..63] scale, [64..127] shift
    const int tid = threadIdx.x;
    if (tid < 64)       ss[tid]      = ws[WS_SCALE_F + tid];
    else if (tid < 128) ss[tid]      = ws[WS_SHIFT_F + (tid - 64)];
    __syncthreads();

    const size_t i4 = (size_t)blockIdx.x * THREADS + tid;  // 2^24 float4s total
    float4* o4 = (float4*)out;
    float4 v = o4[i4];
    const int c0 = (int)(i4 & 15) * 4;   // channel of .x (COUT=64, 16 f4 per row)
    v.x = fminf(fmaxf(fmaf(ss[c0 + 0], v.x, ss[64 + c0 + 0]), 0.f), 6.f);
    v.y = fminf(fmaxf(fmaf(ss[c0 + 1], v.y, ss[64 + c0 + 1]), 0.f), 6.f);
    v.z = fminf(fmaxf(fmaf(ss[c0 + 2], v.z, ss[64 + c0 + 2]), 0.f), 6.f);
    v.w = fminf(fmaxf(fmaf(ss[c0 + 3], v.w, ss[64 + c0 + 3]), 0.f), 6.f);
    o4[i4] = v;
}

// ---------------------------------------------------------------------------
extern "C" void kernel_launch(void* const* d_in, const int* in_sizes, int n_in,
                              void* d_out, int out_size, void* d_ws, size_t ws_size,
                              hipStream_t stream) {
    const float*         feats   = (const float*)d_in[0];
    const float*         W       = (const float*)d_in[1];
    const float*         gamma   = (const float*)d_in[2];
    const float*         beta    = (const float*)d_in[3];
    const int*           nb_idx  = (const int*)d_in[4];
    const unsigned char* nb_mask = (const unsigned char*)d_in[5];
    float* out = (float*)d_out;
    float* ws  = (float*)d_ws;

    (void)in_sizes; (void)n_in; (void)out_size; (void)ws_size;

    prep_kernel<<<1, THREADS, 0, stream>>>(W, ws);
    main_kernel<<<MAIN_BLOCKS, THREADS, 0, stream>>>(feats, nb_idx, nb_mask,
                                                     ws, ws /*partials at base*/, out);
    stats_kernel<<<1, 64, 0, stream>>>(ws, gamma, beta);
    const int norm_blocks = (NPTS * COUT / 4) / THREADS;  // 65536
    norm_kernel<<<norm_blocks, THREADS, 0, stream>>>(ws, out);
}
